// GraphAttentionLayer_86964497809995
// MI455X (gfx1250) — matrix operand order repaired and checked by reading, not compiled, not run on hardware
//
#include <hip/hip_runtime.h>

typedef _Float16 half_t;
typedef __attribute__((ext_vector_type(8)))  _Float16 h8;
typedef __attribute__((ext_vector_type(16))) _Float16 v16h;
typedef __attribute__((ext_vector_type(8)))  float    v8f;

#define NODES 16     // nodes per workgroup == one full WMMA M-tile (no padding)
#define KNBR  20     // neighbors per node
#define EDIM  256    // q/k/v embedding dim
#define DDIM  128    // feature dim / head dim / output dim
#define BLOCK 512    // 16 wave32
#define NWAVE (BLOCK / 32)

// f16 weight workspace layout (element offsets). For every GEMM we arrange B so
// that B[k][n] = W[n*ldb + k]: each lane reads two 8-half runs of one W row.
#define OFF_WQ   0         // [256][256] Wq as-is
#define OFF_WKT  65536     // [256][256] wkT[c][e] = Wk[e][c] (transposed)
#define OFF_WV   131072    // [256][256] Wv as-is
#define OFF_WO   196608    // [256][256] Wo as-is
#define OFF_W1   262144    // [128][384] W1 as-is
#define OFF_W2   311296    // [128][128] W2 as-is

__global__ __launch_bounds__(256)
void cvt_weights(const float* __restrict__ Wq, const float* __restrict__ Wk,
                 const float* __restrict__ Wv, const float* __restrict__ Wo,
                 const float* __restrict__ W1, const float* __restrict__ W2,
                 half_t* __restrict__ w) {
  int i = blockIdx.x * 256 + threadIdx.x;
  if (i < 65536) {
    w[OFF_WQ + i] = (half_t)Wq[i];
    w[OFF_WV + i] = (half_t)Wv[i];
    w[OFF_WO + i] = (half_t)Wo[i];
    int e = i >> 8, c = i & 255;                 // Wk[e][c]
    w[OFF_WKT + c * 256 + e] = (half_t)Wk[i];    // transpose
  }
  if (i < 49152) w[OFF_W1 + i] = (half_t)W1[i];
  if (i < 16384) w[OFF_W2 + i] = (half_t)W2[i];
}

static __device__ __forceinline__ v16h combine(h8 lo, h8 hi) {
  v16h r;
#pragma unroll
  for (int i = 0; i < 8; ++i) { r[i] = lo[i]; r[8 + i] = hi[i]; }
  return r;
}

// One 16x16 output tile (columns n0..n0+15) of C = A(16 x KC) * B(KC x Ntot),
// A in LDS (row stride lda halfs), B given as W rows: B[k][n] = W[n*ldb + k].
// Operand layouts per CDNA5 ISA 7.12.2 (wave32):
//   A: lane m=lane&15 holds row m; halfs = A[m][k+b..+7], A[m][k+16+b..+7], b=(lane>>4)*8
//   B: lane holds column n0+(lane&15); halfs = B[k+b..+7][n], B[k+16+b..+7][n], b=(lane>>4)*8
static __device__ __forceinline__ v8f wmma_tile(const half_t* __restrict__ As, int lda,
                                                const half_t* __restrict__ W, int ldb,
                                                int n0, int KC, int lane) {
  const half_t* arow = As + (lane & 15) * lda + ((lane >> 4) << 3);
  const half_t* brow = W + (n0 + (lane & 15)) * ldb + ((lane >> 4) << 3);
  v8f acc = {};
  for (int k = 0; k < KC; k += 32) {
    v16h a = combine(*(const h8*)(arow + k), *(const h8*)(arow + k + 16));
    v16h b = combine(*(const h8*)(brow + k), *(const h8*)(brow + k + 16));
    acc = __builtin_amdgcn_wmma_f32_16x16x32_f16(false, a, false, b, (short)0, acc,
                                                 false, false);
  }
  return acc;
}

__global__ __launch_bounds__(BLOCK)
void gat_fused(const float* __restrict__ q, const float* __restrict__ qt,
               const float* __restrict__ kv, const float* __restrict__ kvt,
               const int* __restrict__ mask,
               const float* __restrict__ bq, const float* __restrict__ bk,
               const float* __restrict__ bv, const float* __restrict__ bo,
               const float* __restrict__ b1, const float* __restrict__ b2,
               const half_t* __restrict__ w, float* __restrict__ out, int N) {
  // ~207 KB LDS -> 1 workgroup (16 waves) per 320 KB WGP
  __shared__ __align__(16) half_t sKVC[NODES * KNBR * EDIM];  // 160 KB, read once from HBM
  __shared__ __align__(16) half_t bufA[16 * EDIM];            // 8 KB : qc -> o
  __shared__ __align__(16) half_t bufB[16 * 2 * EDIM];        // 16 KB: qh -> m
  __shared__ __align__(16) half_t bufC[16 * 2 * EDIM];        // 16 KB: u  -> h1[16][384]
  __shared__ __align__(16) half_t sH2[16 * DDIM];             // 4 KB
  __shared__ float sAttn[NODES * 2 * KNBR];                   // 2.5 KB
  __shared__ float sCst[NODES * 2];
  __shared__ int   sNoN[NODES];

  const int tid   = threadIdx.x;
  const int wave  = tid >> 5;
  const int lane  = tid & 31;
  const int node0 = blockIdx.x * NODES;
  const int mBase = (lane >> 4) << 3;   // C/D: rows mBase..mBase+7 in this lane half
  const int nLane = lane & 15;          // C/D: column within tile

  // ---- phase 0: stage kvc = concat(kv, kv_time) and qc = concat(q, q_time) as f16
  for (int i = tid; i < NODES * KNBR * 64; i += BLOCK) {   // float4 granules
    int row = i >> 6, j = i & 63;
    int nl = row / KNBR, kk = row - nl * KNBR;
    int g = node0 + nl; if (g >= N) g = N - 1;
    const float* src = (j < 32) ? kv : kvt;
    long base = ((long)g * KNBR + kk) * DDIM;
    float4 v = ((const float4*)(src + base))[j & 31];
    half_t* d = &sKVC[row * EDIM + j * 4];
    d[0] = (half_t)v.x; d[1] = (half_t)v.y; d[2] = (half_t)v.z; d[3] = (half_t)v.w;
  }
  for (int i = tid; i < 16 * 64; i += BLOCK) {
    int row = i >> 6, j = i & 63;
    half_t* d = &bufA[row * EDIM + j * 4];
    if (node0 + row < N) {
      const float* src = (j < 32) ? q : qt;
      long base = (long)(node0 + row) * DDIM;
      float4 v = ((const float4*)(src + base))[j & 31];
      d[0] = (half_t)v.x; d[1] = (half_t)v.y; d[2] = (half_t)v.z; d[3] = (half_t)v.w;
    } else {
      d[0] = d[1] = d[2] = d[3] = (half_t)0.f;
    }
  }
  if (tid < NODES) {
    int g = node0 + tid; if (g >= N) g = N - 1;
    const int* mp = mask + (long)g * KNBR;
    int s = 0;
    for (int k = 0; k < KNBR; ++k) s += (mp[k] != 0);
    sNoN[tid] = (s == 0);
  }
  __syncthreads();

  // ---- phase 1: qh = qc @ Wq^T + bq   -> bufB[16][256]
  for (int t = wave; t < 16; t += NWAVE) {
    int n0 = t << 4;
    v8f acc = wmma_tile(bufA, EDIM, w + OFF_WQ, EDIM, n0, EDIM, lane);
    float bias = bq[n0 + nLane];
#pragma unroll
    for (int r = 0; r < 8; ++r)
      bufB[(mBase + r) * EDIM + n0 + nLane] = (half_t)(acc[r] + bias);
  }
  __syncthreads();

  // ---- phase 2: u_h = qh_h @ Wk_h (query pushed through key projection) -> bufC[16][512]
  for (int t = wave; t < 32; t += NWAVE) {
    int h = t >> 4, n0 = (t & 15) << 4;
    v8f acc = wmma_tile(bufB + h * DDIM, EDIM,
                        w + OFF_WKT + h * DDIM, EDIM, n0, DDIM, lane);
#pragma unroll
    for (int r = 0; r < 8; ++r)
      bufC[(mBase + r) * 2 * EDIM + h * EDIM + n0 + nLane] = (half_t)acc[r];
  }
  // constant score term: qh_h . bk_h
  if (tid < NODES * 2) {
    int nl = tid >> 1, h = tid & 1;
    float s = 0.f;
    for (int d = 0; d < DDIM; ++d)
      s += (float)bufB[nl * EDIM + h * DDIM + d] * bk[h * DDIM + d];
    sCst[tid] = s;
  }
  __syncthreads();

  // ---- phase 3: scores[nl][h][k] = (kvc_k . u_h + qh.bk) / sqrt(128), masked
  for (int i = tid; i < NODES * 2 * KNBR; i += BLOCK) {
    int nl = i / (2 * KNBR);
    int rem = i - nl * 2 * KNBR;
    int h = rem / KNBR, k = rem - h * KNBR;
    const h8* kr = (const h8*)&sKVC[(nl * KNBR + k) * EDIM];
    const h8* ur = (const h8*)&bufC[nl * 2 * EDIM + h * EDIM];
    float s = 0.f;
    for (int c = 0; c < EDIM / 8; ++c) {
      h8 a = kr[c], b = ur[c];
#pragma unroll
      for (int j = 0; j < 8; ++j) s += (float)a[j] * (float)b[j];
    }
    s = (s + sCst[nl * 2 + h]) * 0.08838834764831845f;  // 1/sqrt(128)
    int g = node0 + nl; if (g >= N) g = N - 1;
    bool valid = (mask[(long)g * KNBR + k] != 0) || sNoN[nl];  // no-neigh rows unmasked
    sAttn[i] = valid ? s : -1e30f;
  }
  __syncthreads();

  // ---- phase 4: softmax over k for each (node, head)
  if (tid < NODES * 2) {
    float* row = &sAttn[tid * KNBR];
    float m = row[0];
    for (int k = 1; k < KNBR; ++k) m = fmaxf(m, row[k]);
    float ssum = 0.f;
    for (int k = 0; k < KNBR; ++k) { float e = __expf(row[k] - m); row[k] = e; ssum += e; }
    float inv = 1.f / ssum;
    for (int k = 0; k < KNBR; ++k) row[k] *= inv;
  }
  __syncthreads();

  // ---- phase 5: m_h = attn_h @ kvc (softmax sums to 1 => bv folds out) -> bufB[16][512]
  for (int i = tid; i < NODES * 2 * (EDIM / 8); i += BLOCK) {  // (node, head, 8-col chunk)
    int nl = i >> 6, rem = i & 63, h = rem >> 5, c8 = rem & 31;
    const float* a = &sAttn[(nl * 2 + h) * KNBR];
    const h8* base = (const h8*)&sKVC[nl * KNBR * EDIM] + c8;  // k-stride = 32 h8
    float acc[8];
#pragma unroll
    for (int j = 0; j < 8; ++j) acc[j] = 0.f;
    for (int k = 0; k < KNBR; ++k) {
      h8 v = base[k * (EDIM / 8)];
      float wk = a[k];
#pragma unroll
      for (int j = 0; j < 8; ++j) acc[j] += wk * (float)v[j];
    }
    h8 o;
#pragma unroll
    for (int j = 0; j < 8; ++j) o[j] = (half_t)acc[j];
    *(h8*)&bufB[nl * 2 * EDIM + h * EDIM + c8 * 8] = o;
  }
  __syncthreads();

  // ---- phase 6: o_h = m_h @ Wv_h^T + bv  -> bufA[16][256]
  for (int t = wave; t < 16; t += NWAVE) {
    int h = t >> 3, n0 = (t & 7) << 4;
    v8f acc = wmma_tile(bufB + h * EDIM, 2 * EDIM,
                        w + OFF_WV + (h * DDIM) * EDIM, EDIM, n0, EDIM, lane);
    int e = h * DDIM + n0 + nLane;
    float bias = bv[e];
#pragma unroll
    for (int r = 0; r < 8; ++r)
      bufA[(mBase + r) * EDIM + e] = (half_t)(acc[r] + bias);
  }
  __syncthreads();

  // ---- phase 7: y = o @ Wo^T + bo (zero no-neighbor rows) -> h1 = [q | y] in bufC[16][384]
  for (int t = wave; t < 16; t += NWAVE) {
    int n0 = t << 4;
    v8f acc = wmma_tile(bufA, EDIM, w + OFF_WO, EDIM, n0, EDIM, lane);
    int n = n0 + nLane;
    float bias = bo[n];
#pragma unroll
    for (int r = 0; r < 8; ++r) {
      int m = mBase + r;
      float v = acc[r] + bias;
      if (sNoN[m]) v = 0.f;
      bufC[m * 384 + DDIM + n] = (half_t)v;
    }
  }
  for (int i = tid; i < 16 * DDIM; i += BLOCK) {   // h1 cols 0..127 = src features q
    int row = i >> 7, c = i & 127;
    float v = 0.f;
    if (node0 + row < N) v = q[(long)(node0 + row) * DDIM + c];
    bufC[row * 384 + c] = (half_t)v;
  }
  __syncthreads();

  // ---- phase 8: h2 = relu(h1 @ W1^T + b1) -> sH2[16][128]
  for (int t = wave; t < 8; t += NWAVE) {
    int n0 = t << 4;
    v8f acc = wmma_tile(bufC, 384, w + OFF_W1, 384, n0, 384, lane);
    float bias = b1[n0 + nLane];
#pragma unroll
    for (int r = 0; r < 8; ++r)
      sH2[(mBase + r) * DDIM + n0 + nLane] = (half_t)fmaxf(acc[r] + bias, 0.f);
  }
  __syncthreads();

  // ---- phase 9: out = h2 @ W2^T + b2 (f32 store, rows past N are not stored)
  for (int t = wave; t < 8; t += NWAVE) {
    int n0 = t << 4;
    v8f acc = wmma_tile(sH2, DDIM, w + OFF_W2, DDIM, n0, DDIM, lane);
    float bias = b2[n0 + nLane];
#pragma unroll
    for (int r = 0; r < 8; ++r) {
      int g = node0 + mBase + r;
      if (g < N) out[(long)g * DDIM + n0 + nLane] = acc[r] + bias;
    }
  }
}

extern "C" void kernel_launch(void* const* d_in, const int* in_sizes, int n_in,
                              void* d_out, int out_size, void* d_ws, size_t ws_size,
                              hipStream_t stream) {
  const float* q    = (const float*)d_in[0];
  const float* qt   = (const float*)d_in[1];
  const float* kv   = (const float*)d_in[2];
  const float* kvt  = (const float*)d_in[3];
  const int*   mask = (const int*)d_in[4];
  const float* Wq = (const float*)d_in[5];
  const float* bq = (const float*)d_in[6];
  const float* Wk = (const float*)d_in[7];
  const float* bk = (const float*)d_in[8];
  const float* Wv = (const float*)d_in[9];
  const float* bv = (const float*)d_in[10];
  const float* Wo = (const float*)d_in[11];
  const float* bo = (const float*)d_in[12];
  const float* W1 = (const float*)d_in[13];
  const float* b1 = (const float*)d_in[14];
  const float* W2 = (const float*)d_in[15];
  const float* b2 = (const float*)d_in[16];
  float* out = (float*)d_out;
  half_t* w  = (half_t*)d_ws;   // ~656 KB of f16 weights, L2-resident

  int N = in_sizes[0] / DDIM;   // 20000
  cvt_weights<<<(65536 + 255) / 256, 256, 0, stream>>>(Wq, Wk, Wv, Wo, W1, W2, w);
  int blocks = (N + NODES - 1) / NODES;   // 1250
  gat_fused<<<blocks, BLOCK, 0, stream>>>(q, qt, kv, kvt, mask,
                                          bq, bk, bv, bo, b1, b2, w, out, N);
}
